// GraphLinearDeform_13692355739968
// MI455X (gfx1250) — matrix operand order repaired and checked by reading, not compiled, not run on hardware
//
#include <hip/hip_runtime.h>
#include <hip/hip_bf16.h>
#include <math.h>

typedef __attribute__((ext_vector_type(16))) _Float16 v16h;
typedef __attribute__((ext_vector_type(8)))  float    v8f;

#define NEG_SLOPE 0.3f
#define SCALE_IT  (1.0f / 3.0f)
#define N_ITER    3

#define IMG_C 32
#define IMG_D 96
#define IMG_H 96
#define IMG_W 96
#define DHW   (IMG_D * IMG_H * IMG_W)

#define WAVES_PER_WG 8
#define TPB          (WAVES_PER_WG * 32)
#define PTS_PER_WG   (WAVES_PER_WG * 16)

// ---- LDS layout (dynamic shared) ----
#define SW0_H (128 * 32)
#define SW1_H (128 * 128)
#define SW2_H (64 * 128)
#define WHALVES (SW0_H + SW1_H + SW2_H)          // 28672 halves = 57344 B
#define NFLOATS (128 + 128 + 64 + 192 + 4)       // biases + wd + bd(pad)
#define ACT_HALVES_PER_WAVE (2 * 16 * 128)       // two ping-pong 16x128 f16 bufs
#define SMEM_BYTES (WHALVES * 2 + NFLOATS * 4 + WAVES_PER_WG * ACT_HALVES_PER_WAVE * 2)

__device__ __forceinline__ int clampi(int v, int lo, int hi) {
    return v < lo ? lo : (v > hi ? hi : v);
}

// Wave-local LDS sync: activation buffers are private to one wave, and CDNA5
// keeps a wave's LDS ops in order, so cross-lane visibility within the wave
// only needs DScnt==0 (no workgroup barrier -> waves stay decoupled).
__device__ __forceinline__ void wave_lds_sync() {
    asm volatile("s_wait_dscnt 0x0" ::: "memory");
}

// One MLP layer for a 16-point tile owned by one wave.
//   inAct : LDS f16 [16][CIN]   (row-major, 16B aligned, CIN in {32,128})
//   outAct: LDS f16 [16][COUT]
//   W     : LDS f16 [COUT][CIN] (row-major)
// Computes out[m][n] = leaky( sum_k in[m][k] * W[n][k] + bias[n] )
template <int CIN, int COUT>
__device__ __forceinline__ void mlp_layer(const _Float16* __restrict__ inAct,
                                          _Float16* __restrict__ outAct,
                                          const _Float16* __restrict__ W,
                                          const float* __restrict__ bias,
                                          int lane) {
    constexpr int KC = CIN / 32;  // K chunks of 32
    const int m = lane & 15;      // A-row / C-col index for this lane
    const int g = lane >> 4;      // lane group

    // A fragments (hoisted, reused across all output-column tiles).
    // v16h halves: h<8 -> K = h + 8g ; h>=8 -> K = h + 8 + 8g (pairs = consecutive K)
    v16h afrag[KC];
#pragma unroll
    for (int kc = 0; kc < KC; ++kc) {
        const _Float16* row = inAct + m * CIN + kc * 32;
        v16h a;
        unsigned int* au = reinterpret_cast<unsigned int*>(&a);
#pragma unroll
        for (int v = 0; v < 4; ++v) {
            au[v]     = *reinterpret_cast<const unsigned int*>(row + 2 * v + 8 * g);
            au[4 + v] = *reinterpret_cast<const unsigned int*>(row + 16 + 2 * v + 8 * g);
        }
        afrag[kc] = a;
    }

#pragma unroll
    for (int n0 = 0; n0 < COUT / 16; ++n0) {
        const int n = n0 * 16 + m;     // output column this lane holds
        const float bv = bias[n];
        // Bias folded into the accumulator init (off the WMMA critical path).
        v8f acc = {bv, bv, bv, bv, bv, bv, bv, bv};
#pragma unroll
        for (int kc = 0; kc < KC; ++kc) {
            // B fragment for column n: K = kc*32 + 8g + [0..7], then kc*32 + 16 + 8g + [0..7]
            const _Float16* wrow = W + n * CIN + kc * 32 + g * 8;
            v16h bfr;
            uint4* bu = reinterpret_cast<uint4*>(&bfr);
            bu[0] = *reinterpret_cast<const uint4*>(wrow);
            bu[1] = *reinterpret_cast<const uint4*>(wrow + 16);
            acc = __builtin_amdgcn_wmma_f32_16x16x32_f16(
                false, afrag[kc], false, bfr, (short)0, acc, false, false);
        }
#pragma unroll
        for (int r = 0; r < 8; ++r) {   // VGPR r -> row (r + 8g)
            // LeakyReLU (slope<1):  leaky(x) == max(x, slope*x)  -> single v_max
            const float x = fmaxf(acc[r], NEG_SLOPE * acc[r]);
            outAct[(r + 8 * g) * COUT + n] = (_Float16)x;
        }
    }
}

__global__ __launch_bounds__(TPB)
void deform_kernel(const float* __restrict__ image,   // (32,96,96,96)
                   const float* __restrict__ verts,   // (M,3)
                   const float* __restrict__ w0, const float* __restrict__ b0,
                   const float* __restrict__ w1, const float* __restrict__ b1,
                   const float* __restrict__ w2, const float* __restrict__ b2,
                   const float* __restrict__ wd, const float* __restrict__ bd,
                   float* __restrict__ out, int M) {
    extern __shared__ char smem[];
    _Float16* sW0 = reinterpret_cast<_Float16*>(smem);
    _Float16* sW1 = sW0 + SW0_H;
    _Float16* sW2 = sW1 + SW1_H;
    float* sB0 = reinterpret_cast<float*>(sW2 + SW2_H);
    float* sB1 = sB0 + 128;
    float* sB2 = sB1 + 128;
    float* sWd = sB2 + 64;    // (3,64) row-major
    float* sBd = sWd + 192;   // 3 (+1 pad)
    _Float16* sAct = reinterpret_cast<_Float16*>(sBd + 4);

    const int tid = threadIdx.x;

    // ---- Stage weights into LDS as f16 (once per workgroup) ----
    for (int i = tid; i < SW0_H; i += TPB) sW0[i] = (_Float16)w0[i];
    for (int i = tid; i < SW1_H; i += TPB) sW1[i] = (_Float16)w1[i];
    for (int i = tid; i < SW2_H; i += TPB) sW2[i] = (_Float16)w2[i];
    if (tid < 128) { sB0[tid] = b0[tid]; sB1[tid] = b1[tid]; }
    if (tid < 64)  { sB2[tid] = b2[tid]; }
    if (tid < 192) { sWd[tid] = wd[tid]; }
    if (tid < 3)   { sBd[tid] = bd[tid]; }
    __syncthreads();   // the only workgroup-wide barrier

    const int wave    = tid >> 5;
    const int lane    = tid & 31;
    const int m_local = lane & 15;   // point within the wave's 16-point tile
    const int cgrp    = lane >> 4;   // which 16-channel half this lane samples

    _Float16* bufA = sAct + wave * ACT_HALVES_PER_WAVE;  // 16x128 f16
    _Float16* bufB = bufA + 16 * 128;                    // 16x128 f16

    const int mg = blockIdx.x * PTS_PER_WG + wave * 16 + m_local;
    const int mc = (mg < M) ? mg : (M - 1);  // clamp: WMMA needs full EXEC

    float vx = verts[3 * mc + 0];
    float vy = verts[3 * mc + 1];
    float vz = verts[3 * mc + 2];

    for (int it = 0; it < N_ITER; ++it) {
        // ---- Phase 1: trilinear sample (border clamp, align_corners=True) ----
        const float fx = (vx + 1.0f) * 0.5f * (IMG_W - 1);
        const float fy = (vy + 1.0f) * 0.5f * (IMG_H - 1);
        const float fz = (vz + 1.0f) * 0.5f * (IMG_D - 1);
        const float flx = floorf(fx), fly = floorf(fy), flz = floorf(fz);
        const float wx = fx - flx, wy = fy - fly, wz = fz - flz;
        const int x0 = clampi((int)flx, 0, IMG_W - 1), x1 = clampi(x0 + 1, 0, IMG_W - 1);
        const int y0 = clampi((int)fly, 0, IMG_H - 1), y1 = clampi(y0 + 1, 0, IMG_H - 1);
        const int z0 = clampi((int)flz, 0, IMG_D - 1), z1 = clampi(z0 + 1, 0, IMG_D - 1);
        const float omx = 1.0f - wx, omy = 1.0f - wy, omz = 1.0f - wz;
        const float c000 = omz * omy * omx, c001 = omz * omy * wx;
        const float c010 = omz * wy * omx,  c011 = omz * wy * wx;
        const float c100 = wz * omy * omx,  c101 = wz * omy * wx;
        const float c110 = wz * wy * omx,   c111 = wz * wy * wx;
        const int b00 = (z0 * IMG_H + y0) * IMG_W;
        const int b01 = (z0 * IMG_H + y1) * IMG_W;
        const int b10 = (z1 * IMG_H + y0) * IMG_W;
        const int b11 = (z1 * IMG_H + y1) * IMG_W;

        const float* pc = image + (size_t)(cgrp * 16) * DHW;
#pragma unroll
        for (int cc = 0; cc < 16; ++cc) {
            const float s = pc[b00 + x0] * c000 + pc[b00 + x1] * c001
                          + pc[b01 + x0] * c010 + pc[b01 + x1] * c011
                          + pc[b10 + x0] * c100 + pc[b10 + x1] * c101
                          + pc[b11 + x0] * c110 + pc[b11 + x1] * c111;
            bufA[m_local * 32 + (cgrp * 16 + cc)] = (_Float16)s;  // X: 16x32
            pc += DHW;
        }
        wave_lds_sync();

        // ---- Phase 2: MLP via WMMA (f16 in, f32 accumulate) ----
        mlp_layer<32, 128>(bufA, bufB, sW0, sB0, lane);   // X(16x32)   -> A0(16x128)
        wave_lds_sync();
        mlp_layer<128, 128>(bufB, bufA, sW1, sB1, lane);  // A0(16x128) -> A1(16x128)
        wave_lds_sync();
        mlp_layer<128, 64>(bufA, bufB, sW2, sB2, lane);   // A1(16x128) -> A2(16x64)
        wave_lds_sync();

        // ---- Phase 3: 64 -> 3 head + vertex update (all lanes, wave-local) ----
        const _Float16* a2 = bufB + m_local * 64;
        float d0 = sBd[0], d1 = sBd[1], d2 = sBd[2];
#pragma unroll
        for (int k = 0; k < 64; ++k) {
            const float a = (float)a2[k];
            d0 += sWd[k] * a;
            d1 += sWd[64 + k] * a;
            d2 += sWd[128 + k] * a;
        }
        vx += SCALE_IT * d0;
        vy += SCALE_IT * d1;
        vz += SCALE_IT * d2;
    }

    if (cgrp == 0 && mg < M) {
        out[3 * mg + 0] = vx;
        out[3 * mg + 1] = vy;
        out[3 * mg + 2] = vz;
    }
}

extern "C" void kernel_launch(void* const* d_in, const int* in_sizes, int n_in,
                              void* d_out, int out_size, void* d_ws, size_t ws_size,
                              hipStream_t stream) {
    const float* image = (const float*)d_in[0];
    const float* verts = (const float*)d_in[1];
    const float* w0 = (const float*)d_in[2];
    const float* b0 = (const float*)d_in[3];
    const float* w1 = (const float*)d_in[4];
    const float* b1 = (const float*)d_in[5];
    const float* w2 = (const float*)d_in[6];
    const float* b2 = (const float*)d_in[7];
    const float* wd = (const float*)d_in[8];
    const float* bd = (const float*)d_in[9];

    const int M = in_sizes[1] / 3;
    const int blocks = (M + PTS_PER_WG - 1) / PTS_PER_WG;

    // Allow > default dynamic LDS (gfx1250 WGP has 320 KB).
    (void)hipFuncSetAttribute((const void*)deform_kernel,
                              hipFuncAttributeMaxDynamicSharedMemorySize, SMEM_BYTES);

    deform_kernel<<<blocks, TPB, SMEM_BYTES, stream>>>(
        image, verts, w0, b0, w1, b1, w2, b2, wd, bd, (float*)d_out, M);
}
